// Attention_50886772523288
// MI455X (gfx1250) — matrix operand order repaired and checked by reading, not compiled, not run on hardware
//
#include <hip/hip_runtime.h>
#include <hip/hip_bf16.h>

#define DIM 4096
#define NHEADS 32
#define HD 128
#define BSZ 2
#define SEQ 2048
#define NTOK (BSZ * SEQ)   // 4096 tokens

typedef __attribute__((ext_vector_type(16))) __bf16 bf16x16;
typedef __attribute__((ext_vector_type(4)))  __bf16 bf16x4;
typedef __attribute__((ext_vector_type(8)))  float  f32x8;
typedef __attribute__((ext_vector_type(4)))  unsigned int u32x4;

union FragU {
  u32x4   q[2];
  bf16x16 v;
};

__device__ __forceinline__ f32x8 wmma_bf16(bf16x16 a, bf16x16 b, f32x8 c) {
  // D = A(16x32 bf16) * B(32x16 bf16) + C(16x16 f32)
  return __builtin_amdgcn_wmma_f32_16x16x32_bf16(false, a, false, b, (short)0, c,
                                                 false, false);
}

// ---------------------------------------------------------------------------
// fp32 -> bf16 cast, 4 elements / thread
// ---------------------------------------------------------------------------
__global__ __launch_bounds__(256) void cast_f32_bf16_kernel(
    const float* __restrict__ in, __bf16* __restrict__ out) {
  size_t i = ((size_t)blockIdx.x * 256 + threadIdx.x) * 4;
  float4 f = *(const float4*)(in + i);
  bf16x4 o;
  o[0] = (__bf16)f.x; o[1] = (__bf16)f.y; o[2] = (__bf16)f.z; o[3] = (__bf16)f.w;
  *(bf16x4*)(out + i) = o;
}

// ---------------------------------------------------------------------------
// C[M,N] (f32) = A[M,K] (bf16, row-major) * B[N,K]^T (bf16, row-major rows = N)
// Block: 256 threads = 8 waves. Wave tile: 32(M) x 64(N) -> 8 WMMA per
// 6 fragment loads (2 A-frags amortize 4 B-frags). Block tile: 64 x 256.
// All fragment loads are contiguous b128 per the wave32 WMMA layouts.
// ---------------------------------------------------------------------------
__global__ __launch_bounds__(256) void gemm_bf16_nt_kernel(
    const __bf16* __restrict__ A, const __bf16* __restrict__ B,
    float* __restrict__ C, int M, int N, int K) {
  const int lane = threadIdx.x & 31;
  const int wave = threadIdx.x >> 5;
  const int lh   = lane & 15;
  const int sel  = lane >> 4;                       // half-wave selector
  const int mbase = blockIdx.y * 64 + (wave & 1) * 32;
  const int nbase = blockIdx.x * 256 + (wave >> 1) * 64;

  const __bf16* arow0 = A + (size_t)(mbase + 0 + lh) * K;
  const __bf16* arow1 = A + (size_t)(mbase + 16 + lh) * K;
  const __bf16* br0  = B + (size_t)(nbase + 0 + lh) * K;
  const __bf16* br1  = B + (size_t)(nbase + 16 + lh) * K;
  const __bf16* br2  = B + (size_t)(nbase + 32 + lh) * K;
  const __bf16* br3  = B + (size_t)(nbase + 48 + lh) * K;

  f32x8 acc00 = {}, acc01 = {}, acc02 = {}, acc03 = {};
  f32x8 acc10 = {}, acc11 = {}, acc12 = {}, acc13 = {};

  for (int k0 = 0; k0 < K; k0 += 32) {
    FragU fa0, fa1;
    fa0.q[0] = *(const u32x4*)(arow0 + k0 + sel * 16);      // K +0..7 / +16..23
    fa0.q[1] = *(const u32x4*)(arow0 + k0 + sel * 16 + 8);  // K +8..15 / +24..31
    fa1.q[0] = *(const u32x4*)(arow1 + k0 + sel * 16);
    fa1.q[1] = *(const u32x4*)(arow1 + k0 + sel * 16 + 8);
    __builtin_prefetch(arow0 + k0 + 64, 0, 1);              // global_prefetch_b8
    __builtin_prefetch(arow1 + k0 + 64, 0, 1);

    FragU fb;
    fb.q[0] = *(const u32x4*)(br0 + k0 + sel * 16);
    fb.q[1] = *(const u32x4*)(br0 + k0 + sel * 16 + 8);
    acc00 = wmma_bf16(fa0.v, fb.v, acc00);
    acc10 = wmma_bf16(fa1.v, fb.v, acc10);

    fb.q[0] = *(const u32x4*)(br1 + k0 + sel * 16);
    fb.q[1] = *(const u32x4*)(br1 + k0 + sel * 16 + 8);
    acc01 = wmma_bf16(fa0.v, fb.v, acc01);
    acc11 = wmma_bf16(fa1.v, fb.v, acc11);

    fb.q[0] = *(const u32x4*)(br2 + k0 + sel * 16);
    fb.q[1] = *(const u32x4*)(br2 + k0 + sel * 16 + 8);
    acc02 = wmma_bf16(fa0.v, fb.v, acc02);
    acc12 = wmma_bf16(fa1.v, fb.v, acc12);

    fb.q[0] = *(const u32x4*)(br3 + k0 + sel * 16);
    fb.q[1] = *(const u32x4*)(br3 + k0 + sel * 16 + 8);
    acc03 = wmma_bf16(fa0.v, fb.v, acc03);
    acc13 = wmma_bf16(fa1.v, fb.v, acc13);
    __builtin_prefetch(br3 + k0 + 64, 0, 1);
  }

  // D layout: lane -> N = nt*16+lh ; vgpr r -> M = mtile + sel*8 + r
  float* crow0 = C + (size_t)(mbase + 0 + sel * 8) * N + nbase + lh;
  float* crow1 = C + (size_t)(mbase + 16 + sel * 8) * N + nbase + lh;
#pragma unroll
  for (int r = 0; r < 8; ++r) {
    crow0[(size_t)r * N + 0]  = acc00[r];
    crow0[(size_t)r * N + 16] = acc01[r];
    crow0[(size_t)r * N + 32] = acc02[r];
    crow0[(size_t)r * N + 48] = acc03[r];
    crow1[(size_t)r * N + 0]  = acc10[r];
    crow1[(size_t)r * N + 16] = acc11[r];
    crow1[(size_t)r * N + 32] = acc12[r];
    crow1[(size_t)r * N + 48] = acc13[r];
  }
}

// ---------------------------------------------------------------------------
// RoPE on q,k (fp32 GEMM outputs, [tok, DIM]) + KV cache write (fp32, d_out)
// + bf16 repack: q/k -> [b][h][s][d], v -> transposed [b][h][d][s].
// One thread per (b, s, h, pair i).
// ---------------------------------------------------------------------------
__global__ __launch_bounds__(256) void rope_cache_kernel(
    const float* __restrict__ qf, const float* __restrict__ kf,
    const float* __restrict__ vf, const float* __restrict__ freqs,
    float* __restrict__ cache, __bf16* __restrict__ qb,
    __bf16* __restrict__ kb, __bf16* __restrict__ vtb) {
  int idx = blockIdx.x * 256 + threadIdx.x;
  int i = idx & 63;            // rotary pair
  int h = (idx >> 6) & 31;
  int s = (idx >> 11) & 2047;
  int b = idx >> 22;

  size_t base = ((size_t)b * SEQ + s) * DIM + h * HD + 2 * i;
  float c  = freqs[(size_t)s * 128 + 2 * i + 0];
  float sn = freqs[(size_t)s * 128 + 2 * i + 1];

  float q0 = qf[base], q1 = qf[base + 1];
  float k0 = kf[base], k1 = kf[base + 1];
  float v0 = vf[base], v1 = vf[base + 1];

  float qr = q0 * c - q1 * sn, qi = q0 * sn + q1 * c;
  float kr = k0 * c - k1 * sn, ki = k0 * sn + k1 * c;

  // new_hidden: [2][BSZ][SEQ][NHEADS][HD]
  size_t coff = (((size_t)b * SEQ + s) * NHEADS + h) * HD + 2 * i;
  const size_t plane = (size_t)BSZ * SEQ * NHEADS * HD;
  cache[coff]             = kr;  cache[coff + 1]         = ki;
  cache[plane + coff]     = v0;  cache[plane + coff + 1] = v1;

  size_t qoff = (((size_t)b * NHEADS + h) * SEQ + s) * HD + 2 * i;
  qb[qoff] = (__bf16)qr; qb[qoff + 1] = (__bf16)qi;
  kb[qoff] = (__bf16)kr; kb[qoff + 1] = (__bf16)ki;

  size_t voff = (((size_t)b * NHEADS + h) * HD + 2 * i) * SEQ + s;
  vtb[voff]       = (__bf16)v0;
  vtb[voff + SEQ] = (__bf16)v1;
}

// ---------------------------------------------------------------------------
// Flash attention, one wave per (b, h, 16-row q tile).
// S^T = K·Q^T via WMMA (rows=k, cols=q) so P converts in-lane to the P·V
// A-fragment; V is pre-transposed so the P·V B-fragment is contiguous.
// ---------------------------------------------------------------------------
__global__ __launch_bounds__(32) void flash_attn_kernel(
    const __bf16* __restrict__ qb, const __bf16* __restrict__ kb,
    const __bf16* __restrict__ vtb, __bf16* __restrict__ attnb) {
  const int lane = threadIdx.x & 31;
  const int lh = lane & 15, sel = lane >> 4;
  const int qbase = blockIdx.x * 16;
  const int h = blockIdx.y, b = blockIdx.z;

  const __bf16* Q  = qb  + ((size_t)b * NHEADS + h) * SEQ * HD;
  const __bf16* Kp = kb  + ((size_t)b * NHEADS + h) * SEQ * HD;
  const __bf16* Vt = vtb + ((size_t)b * NHEADS + h) * HD * SEQ;

  // Q^T B-fragments for the 4 d-chunks of 32 (lane -> q column, K = d)
  bf16x16 qfrag[4];
#pragma unroll
  for (int dc = 0; dc < 4; ++dc) {
    FragU f;
    const __bf16* p = Q + (size_t)(qbase + lh) * HD + dc * 32 + sel * 8;
    f.q[0] = *(const u32x4*)p;
    f.q[1] = *(const u32x4*)(p + 16);
    qfrag[dc] = f.v;
  }

  f32x8 o[8] = {{0}};              // O tiles: 8 x (16q x 16d), D layout
  float m = -3.0e38f, l = 0.0f;
  const float scale = 0.08838834764831845f;   // 1/sqrt(128)
  const int q_global = qbase + lh;
  const int nk = qbase + 16;                  // causal bound (exclusive)

  for (int kb0 = 0; kb0 < nk; kb0 += 32) {
    // --- S^T tiles: rows k in [kb0,kb0+16) and [kb0+16,kb0+32) ---
    f32x8 s0 = {}, s1 = {};
#pragma unroll
    for (int dc = 0; dc < 4; ++dc) {
      FragU fa;
      const __bf16* p0 = Kp + (size_t)(kb0 + lh) * HD + dc * 32 + sel * 8;
      fa.q[0] = *(const u32x4*)p0;
      fa.q[1] = *(const u32x4*)(p0 + 16);
      s0 = wmma_bf16(fa.v, qfrag[dc], s0);
      const __bf16* p1 = Kp + (size_t)(kb0 + 16 + lh) * HD + dc * 32 + sel * 8;
      fa.q[0] = *(const u32x4*)p1;
      fa.q[1] = *(const u32x4*)(p1 + 16);
      s1 = wmma_bf16(fa.v, qfrag[dc], s1);
    }

    // --- scale + causal mask + online softmax (lane owns q = qbase+lh) ---
    float p0v[8], p1v[8];
    float mnew = m;
#pragma unroll
    for (int r = 0; r < 8; ++r) {
      int k0i = kb0 + r + 8 * sel;
      int k1i = kb0 + 16 + r + 8 * sel;
      float a = s0[r] * scale + ((k0i > q_global) ? -1.0e9f : 0.0f);
      float bb = s1[r] * scale + ((k1i > q_global) ? -1.0e9f : 0.0f);
      p0v[r] = a; p1v[r] = bb;
      mnew = fmaxf(mnew, fmaxf(a, bb));
    }
    mnew = fmaxf(mnew, __shfl_xor(mnew, 16, 32));
    float corr = __expf(m - mnew);
    float rowsum = 0.0f;
#pragma unroll
    for (int r = 0; r < 8; ++r) {
      p0v[r] = __expf(p0v[r] - mnew);
      p1v[r] = __expf(p1v[r] - mnew);
      rowsum += p0v[r] + p1v[r];
    }
    rowsum += __shfl_xor(rowsum, 16, 32);
    l = l * corr + rowsum;
    m = mnew;

    // rescale O accumulators: factor indexed by q = r + 8*sel of the D tile
#pragma unroll
    for (int r = 0; r < 8; ++r) {
      float cr = __shfl(corr, r + 8 * sel, 32);
#pragma unroll
      for (int dt = 0; dt < 8; ++dt) o[dt][r] *= cr;
    }

    // --- P A-fragment: in-lane pack of the two S^T tiles ---
    FragU pa;
#pragma unroll
    for (int j = 0; j < 8; ++j) {
      pa.v[j]     = (__bf16)p0v[j];   // K = kb0 + 0..7  (+8 for upper half-wave)
      pa.v[8 + j] = (__bf16)p1v[j];   // K = kb0 + 16..23 (+8 for upper half-wave)
    }

    // --- O += P * V, V^T memory gives contiguous B-fragments ---
#pragma unroll
    for (int dt = 0; dt < 8; ++dt) {
      FragU vb;
      const __bf16* pv = Vt + (size_t)(dt * 16 + lh) * SEQ + kb0 + sel * 8;
      vb.q[0] = *(const u32x4*)pv;
      vb.q[1] = *(const u32x4*)(pv + 16);
      o[dt] = wmma_bf16(pa.v, vb.v, o[dt]);
    }
  }

  // --- normalize + store attn output (bf16 [tok, DIM]) ---
  float invl = 1.0f / l;
#pragma unroll
  for (int r = 0; r < 8; ++r) {
    float inv_r = __shfl(invl, r + 8 * sel, 32);
    int qrow = qbase + r + 8 * sel;
    size_t row = ((size_t)b * SEQ + qrow) * DIM + h * HD + lh;
#pragma unroll
    for (int dt = 0; dt < 8; ++dt)
      attnb[row + dt * 16] = (__bf16)(o[dt][r] * inv_r);
  }
}

// ---------------------------------------------------------------------------
extern "C" void kernel_launch(void* const* d_in, const int* in_sizes, int n_in,
                              void* d_out, int out_size, void* d_ws, size_t ws_size,
                              hipStream_t stream) {
  (void)in_sizes; (void)n_in; (void)out_size; (void)ws_size;
  const float* x  = (const float*)d_in[0];
  const float* fr = (const float*)d_in[1];
  // d_in[2]=mask (causal mask computed in-kernel), d_in[3]=hidden_state
  // (start_pos==0 and full seqlen => cache fully overwritten), d_in[8]=start_pos
  const float* wq = (const float*)d_in[4];
  const float* wk = (const float*)d_in[5];
  const float* wv = (const float*)d_in[6];
  const float* wo = (const float*)d_in[7];

  float* out   = (float*)d_out;
  float* cache = out + (size_t)NTOK * DIM;     // new_hidden region

  // workspace carve-up (static, 256B aligned)
  char* p = (char*)d_ws;
  auto carve = [&](size_t bytes) {
    char* r = p;
    p += (bytes + 255) & ~(size_t)255;
    return r;
  };
  const size_t EB = (size_t)NTOK * DIM;        // 16,777,216 elements
  __bf16* xb    = (__bf16*)carve(EB * 2);
  __bf16* wqb   = (__bf16*)carve(EB * 2);
  __bf16* wkb   = (__bf16*)carve(EB * 2);
  __bf16* wvb   = (__bf16*)carve(EB * 2);
  __bf16* wob   = (__bf16*)carve(EB * 2);
  float*  qf    = (float*)carve(EB * 4);
  float*  kf    = (float*)carve(EB * 4);
  float*  vf    = (float*)carve(EB * 4);
  __bf16* qb2   = (__bf16*)carve(EB * 2);      // [b][h][s][d]
  __bf16* kb2   = (__bf16*)carve(EB * 2);      // [b][h][s][d]
  __bf16* vtb   = (__bf16*)carve(EB * 2);      // [b][h][d][s]
  __bf16* attnb = (__bf16*)carve(EB * 2);      // [tok][DIM]

  const int castBlocks = (int)(EB / 4 / 256);  // 16384
  cast_f32_bf16_kernel<<<castBlocks, 256, 0, stream>>>(x,  xb);
  cast_f32_bf16_kernel<<<castBlocks, 256, 0, stream>>>(wq, wqb);
  cast_f32_bf16_kernel<<<castBlocks, 256, 0, stream>>>(wk, wkb);
  cast_f32_bf16_kernel<<<castBlocks, 256, 0, stream>>>(wv, wvb);
  cast_f32_bf16_kernel<<<castBlocks, 256, 0, stream>>>(wo, wob);

  dim3 ggrid(DIM / 256, NTOK / 64);            // (16, 64)
  gemm_bf16_nt_kernel<<<ggrid, 256, 0, stream>>>(xb, wqb, qf, NTOK, DIM, DIM);
  gemm_bf16_nt_kernel<<<ggrid, 256, 0, stream>>>(xb, wkb, kf, NTOK, DIM, DIM);
  gemm_bf16_nt_kernel<<<ggrid, 256, 0, stream>>>(xb, wvb, vf, NTOK, DIM, DIM);

  const int ropeBlocks = (int)((size_t)NTOK * NHEADS * (HD / 2) / 256); // 32768
  rope_cache_kernel<<<ropeBlocks, 256, 0, stream>>>(qf, kf, vf, fr, cache,
                                                    qb2, kb2, vtb);

  flash_attn_kernel<<<dim3(SEQ / 16, NHEADS, BSZ), 32, 0, stream>>>(qb2, kb2,
                                                                    vtb, attnb);

  gemm_bf16_nt_kernel<<<ggrid, 256, 0, stream>>>(attnb, wob, out, NTOK, DIM, DIM);
}
